// CustomMamba_14156212208278
// MI455X (gfx1250) — matrix operand order repaired and checked by reading, not compiled, not run on hardware
//
#include <hip/hip_runtime.h>
#include <hip/hip_bf16.h>

typedef __attribute__((ext_vector_type(16))) __bf16 v16bf;
typedef __attribute__((ext_vector_type(8)))  float  v8f;
typedef __attribute__((__vector_size__(16))) int v4i32;
typedef unsigned short u16;

#define NL   4      // layers
#define NB   1024   // batch
#define DIN  512
#define DOUT 512
#define DM   1024
#define DSS  16     // state size
#define DCV  4      // conv width
#define DI   2048   // inner dim
#define DTR  64
#define XDBW 96     // DTR + 2*DS

enum { EPI_NONE = 0, EPI_BIAS = 1, EPI_BIAS_SOFTPLUS = 2 };

#if defined(__HIP_DEVICE_COMPILE__) && \
    __has_builtin(__builtin_amdgcn_tensor_load_to_lds) && \
    __has_builtin(__builtin_amdgcn_s_wait_tensorcnt)
#define USE_TDM 1
#else
#define USE_TDM 0
#endif

#if defined(__HIP_DEVICE_COMPILE__) && \
    __has_builtin(__builtin_amdgcn_global_load_async_to_lds_b128) && \
    __has_builtin(__builtin_amdgcn_s_wait_asynccnt)
#define USE_ASYNC 1
#else
#define USE_ASYNC 0
#endif

__device__ __forceinline__ unsigned bf16_rne(float f) {
  union { float f; unsigned u; } c; c.f = f;
  unsigned u = c.u;
  return (u + 0x7FFFu + ((u >> 16) & 1u)) >> 16;
}

__device__ __forceinline__ unsigned pkbf(float lo, float hi) {
#if defined(__HIP_DEVICE_COMPILE__) && __has_builtin(__builtin_amdgcn_cvt_pk_bf16_f32)
  auto r = __builtin_amdgcn_cvt_pk_bf16_f32(lo, hi);   // single v_cvt_pk_bf16_f32
  unsigned u; __builtin_memcpy(&u, &r, sizeof(u));
  return u;
#else
  return bf16_rne(lo) | (bf16_rne(hi) << 16);
#endif
}

#if USE_ASYNC
__device__ __forceinline__ void async_cp16(const u16* g, unsigned* l) {
  __builtin_amdgcn_global_load_async_to_lds_b128(
      (__attribute__((address_space(1))) v4i32*)g,
      (__attribute__((address_space(3))) v4i32*)l, 0, 0);
}
#endif

#if USE_TDM
typedef __attribute__((ext_vector_type(4))) unsigned tdm_v4u;
typedef __attribute__((ext_vector_type(8))) int      tdm_v8i;
typedef __attribute__((ext_vector_type(4))) int      tdm_v4i;

__device__ __forceinline__ unsigned lds_byte_off(void* p) {
  return (unsigned)(size_t)(__attribute__((address_space(3))) void*)p;
}

// One TDM descriptor moves a 128-row x 32-elem bf16 tile into LDS with
// hardware row-OOB (rowsLeft) and LDS padding (16 dwords data + 4 dwords pad
// per row == the 20-dword bank-spread row stride the WMMA fragments expect).
__device__ __forceinline__ void tdm_load_tile(const u16* gsrc, unsigned ldsByte,
                                              int rowsLeft, int colsLeft, int ldElems) {
  const unsigned long long ga = (unsigned long long)(size_t)gsrc;
  const unsigned dim0 = (unsigned)colsLeft;
  const unsigned dim1 = (unsigned)(rowsLeft < 0 ? 0 : rowsLeft);
  tdm_v4u g0;
  g0[0] = 1u;                                                  // count=1, user D#
  g0[1] = ldsByte;                                             // lds_addr
  g0[2] = (unsigned)(ga & 0xFFFFFFFFu);                        // global_addr[31:0]
  g0[3] = ((unsigned)(ga >> 32) & 0x01FFFFFFu) | (2u << 30);   // addr[56:32] | type=2
  tdm_v8i g1;
  g1[0] = (int)((1u << 16)        // data_size = 2 bytes
              | (1u << 20)        // pad_enable
              | (3u << 22)        // pad_interval: 16 dwords
              | (3u << 25));      // pad_amount: 4 dwords
  g1[1] = (int)((dim0 & 0xFFFFu) << 16);                        // tensor_dim0[15:0]
  g1[2] = (int)((dim0 >> 16) | ((dim1 & 0xFFFFu) << 16));       // dim0[31:16]|dim1[15:0]
  g1[3] = (int)((dim1 >> 16) | (32u << 16));                    // dim1[31:16]|tile_dim0=32
  g1[4] = (int)128u;                                            // tile_dim1=128, tile_dim2=0
  g1[5] = (int)(unsigned)ldElems;                               // tensor_dim0_stride[31:0]
  g1[6] = 0;                                                    // stride[47:32]|dim1_stride lo
  g1[7] = 0;
  tdm_v4i z4 = {0, 0, 0, 0};
#if defined(__clang_major__) && __clang_major__ >= 23
  tdm_v8i z8 = {0, 0, 0, 0, 0, 0, 0, 0};
  __builtin_amdgcn_tensor_load_to_lds(g0, g1, z4, z4, z8, 0);
#else
  __builtin_amdgcn_tensor_load_to_lds(g0, g1, z4, z4, 0);
#endif
}
#endif

// ---------------------------------------------------------------------------
// f32 -> packed bf16 bulk converter (8 elems / thread, b128 in, b128 out)
// ---------------------------------------------------------------------------
__global__ __launch_bounds__(256)
void cvt_bf16_kernel(const float* __restrict__ src, unsigned* __restrict__ dst, long n8)
{
  const long i = (long)blockIdx.x * 256 + threadIdx.x;
  if (i >= n8) return;
  const float4* s = (const float4*)(src + i * 8);
  const float4 a = s[0], b = s[1];
  *(uint4*)(dst + i * 4) =
      make_uint4(pkbf(a.x, a.y), pkbf(a.z, a.w), pkbf(b.x, b.y), pkbf(b.z, b.w));
}

// ---------------------------------------------------------------------------
// C[M,N](f32) = A[M,K](bf16) * W[N,K](bf16)^T (+bias)(+softplus)(+bf16 copy)
// 256 threads = 8 wave32, block tile 128x128, wave tile 32x64, K-step 32.
// Tiles staged by TDM (tensor_load_to_lds) when available, else async-to-LDS,
// else plain load/ds_store. Double-buffered LDS for TDM/async; the plain path reuses one buffer.
// ---------------------------------------------------------------------------
__global__ __launch_bounds__(256)
void wmma_gemm_kernel(const u16* __restrict__ A, const u16* __restrict__ W,
                      const float* __restrict__ bias, float* __restrict__ C,
                      u16* __restrict__ C16, int ldc16,
                      int M, int N, int K, int lda, int ldb, int ldc, int epi)
{
  // 128 rows x 16 dwords (32 bf16), row stride 20 dwords: 16B-aligned, bank-spread
  __shared__ __align__(16) unsigned lAs[2][128 * 20];
  __shared__ __align__(16) unsigned lBs[2][128 * 20];

  const int tid   = threadIdx.x;
  const int lane  = tid & 31;
  const int wid   = tid >> 5;
  const int waveM = wid >> 1;   // 0..3
  const int waveN = wid & 1;    // 0..1
  const int mBase = blockIdx.y * 128;
  const int nBase = blockIdx.x * 128;

  v8f acc[2][4];
#pragma unroll
  for (int i = 0; i < 2; ++i)
#pragma unroll
    for (int j = 0; j < 4; ++j)
#pragma unroll
      for (int r = 0; r < 8; ++r) acc[i][j][r] = 0.0f;

#if USE_TDM
  auto stage = [&](int buf, int k0) {            // wave 0 only: 2 descriptors/tile
    tdm_load_tile(A + (size_t)mBase * lda + k0, lds_byte_off(&lAs[buf][0]),
                  M - mBase, K - k0, lda);
    tdm_load_tile(W + (size_t)nBase * ldb + k0, lds_byte_off(&lBs[buf][0]),
                  N - nBase, K - k0, ldb);
  };
#elif USE_ASYNC
  auto stage = [&](int buf, int k0) {
#pragma unroll
    for (int i = 0; i < 2; ++i) {
      const int ch = tid + i * 256;          // 0..511 : 128 rows x 4 b128-chunks
      const int row = ch >> 2, c = ch & 3;   // chunk = 8 bf16 = 4 dwords
      const int ga = mBase + row;
      if (ga < M) async_cp16(A + (size_t)ga * lda + k0 + c * 8, &lAs[buf][row * 20 + c * 4]);
      const int gb = nBase + row;
      if (gb < N) async_cp16(W + (size_t)gb * ldb + k0 + c * 8, &lBs[buf][row * 20 + c * 4]);
    }
  };
#else
  auto stage = [&](int buf, int k0) {
#pragma unroll
    for (int i = 0; i < 2; ++i) {
      const int ch = tid + i * 256;
      const int row = ch >> 2, c = ch & 3;
      {
        const int ga = mBase + row;
        uint4 v = make_uint4(0u, 0u, 0u, 0u);
        if (ga < M) v = *(const uint4*)(A + (size_t)ga * lda + k0 + c * 8);
        *(uint4*)&lAs[buf][row * 20 + c * 4] = v;
      }
      {
        const int gb = nBase + row;
        uint4 v = make_uint4(0u, 0u, 0u, 0u);
        if (gb < N) v = *(const uint4*)(W + (size_t)gb * ldb + k0 + c * 8);
        *(uint4*)&lBs[buf][row * 20 + c * 4] = v;
      }
    }
  };
#endif

  auto compute = [&](int buf) {
    union Frag { uint4 q[2]; v16bf v; };
    v16bf aF[2], bF[4];
    const int ar  = waveM * 32 + (lane & 15);
    const int kbA = (lane >> 4) * 4;          // dword offset: K 0..7 | 8..15
#pragma unroll
    for (int m = 0; m < 2; ++m) {
      const unsigned* p = &lAs[buf][(ar + m * 16) * 20 + kbA];
      Frag u; u.q[0] = *(const uint4*)p; u.q[1] = *(const uint4*)(p + 8);  // +K16..23
      aF[m] = u.v;
    }
    const int br  = waveN * 64 + (lane & 15);
    const int kbB = (lane >> 4) * 4;          // dword offset: K 0..7 | 8..15
#pragma unroll
    for (int n = 0; n < 4; ++n) {
      const unsigned* p = &lBs[buf][(br + n * 16) * 20 + kbB];
      Frag u; u.q[0] = *(const uint4*)p; u.q[1] = *(const uint4*)(p + 8);
      bF[n] = u.v;
    }
#pragma unroll
    for (int m = 0; m < 2; ++m)
#pragma unroll
      for (int n = 0; n < 4; ++n)
        acc[m][n] = __builtin_amdgcn_wmma_f32_16x16x32_bf16(
            false, aF[m], false, bF[n], (short)0, acc[m][n], false, false);
  };

  const int nK = K >> 5;
#if USE_TDM
  if (wid == 0) stage(0, 0);
  for (int ki = 0; ki < nK; ++ki) {
    const int buf = ki & 1;
    if (wid == 0) __builtin_amdgcn_s_wait_tensorcnt(0);  // tile 'buf' landed
    __syncthreads();                                     // publish to all waves
    if (wid == 0 && ki + 1 < nK) stage(buf ^ 1, (ki + 1) << 5);
    compute(buf);
  }
#elif USE_ASYNC
  stage(0, 0);
  for (int ki = 0; ki < nK; ++ki) {
    const int buf = ki & 1;
    __builtin_amdgcn_s_wait_asynccnt(0);   // tile 'buf' landed in LDS
    __syncthreads();                       // all waves' copies visible; prev reads done
    if (ki + 1 < nK) stage(buf ^ 1, (ki + 1) << 5);
    compute(buf);
  }
#else
  for (int ki = 0; ki < nK; ++ki) {
    stage(0, ki << 5);
    __syncthreads();
    compute(0);
    __syncthreads();
  }
#endif

  // epilogue: C layout = lane (N=lane&15), 8 rows per lane, row-half by lane>>4
#pragma unroll
  for (int m = 0; m < 2; ++m) {
    const int rbase = mBase + waveM * 32 + m * 16 + ((lane >> 4) << 3);
#pragma unroll
    for (int n = 0; n < 4; ++n) {
      const int col = nBase + waveN * 64 + n * 16 + (lane & 15);
      if (col >= N) continue;
      const float bv = (epi != EPI_NONE && bias) ? bias[col] : 0.0f;
#pragma unroll
      for (int r = 0; r < 8; ++r) {
        const int row = rbase + r;
        if (row >= M) continue;
        float v = acc[m][n][r] + bv;
        if (epi == EPI_BIAS_SOFTPLUS) v = (v > 20.0f) ? v : log1pf(__expf(v));
        C[(size_t)row * ldc + col] = v;
        if (C16) C16[(size_t)row * ldc16 + col] = (u16)bf16_rne(v);
      }
    }
  }
}

// ---------------------------------------------------------------------------
// residual accumulate + layernorm over DM=1024; emits bf16 (GEMM A operand)
// ---------------------------------------------------------------------------
__global__ __launch_bounds__(256)
void ln_kernel(const float* __restrict__ h, float* __restrict__ res,
               u16* __restrict__ hn16, const float* __restrict__ w,
               const float* __restrict__ b, int addRes)
{
  __shared__ float s1[256], s2[256];
  const int row = blockIdx.x;
  const int tid = threadIdx.x;
  float v[4], sum = 0.0f, sq = 0.0f;
#pragma unroll
  for (int i = 0; i < 4; ++i) {
    const int c = i * 256 + tid;
    float r = h[(size_t)row * DM + c];
    if (addRes) r += res[(size_t)row * DM + c];
    res[(size_t)row * DM + c] = r;
    v[i] = r; sum += r; sq += r * r;
  }
  s1[tid] = sum; s2[tid] = sq;
  __syncthreads();
  for (int s = 128; s > 0; s >>= 1) {
    if (tid < s) { s1[tid] += s1[tid + s]; s2[tid] += s2[tid + s]; }
    __syncthreads();
  }
  const float mu  = s1[0] * (1.0f / DM);
  const float var = s2[0] * (1.0f / DM) - mu * mu;
  const float rs  = rsqrtf(var + 1e-5f);
#pragma unroll
  for (int i = 0; i < 4; ++i) {
    const int c = i * 256 + tid;
    hn16[(size_t)row * DM + c] = (u16)bf16_rne((v[i] - mu) * rs * w[c] + b[c]);
  }
}

// ---------------------------------------------------------------------------
// conv-state shift + depthwise conv + SiLU; new conv state -> d_out
// ---------------------------------------------------------------------------
__global__ __launch_bounds__(256)
void conv_kernel(const float* __restrict__ st_in, const float* __restrict__ xz,
                 const float* __restrict__ convW, const float* __restrict__ convB,
                 float* __restrict__ st_out, float* __restrict__ xs,
                 u16* __restrict__ xs16, int l)
{
  const int idx = blockIdx.x * 256 + threadIdx.x;   // < NB*DI
  const int b = idx >> 11;
  const int d = idx & (DI - 1);
  const size_t sbase = ((size_t)(l * NB + b) * DI + d) * (DCV + DSS);
  const float c1 = st_in[sbase + 1], c2 = st_in[sbase + 2], c3 = st_in[sbase + 3];
  const float xc = xz[(size_t)b * (2 * DI) + d];
  const float* wp = convW + ((size_t)l * DI + d) * DCV;
  float a = c1 * wp[0] + c2 * wp[1] + c3 * wp[2] + xc * wp[3] + convB[l * DI + d];
  const float s = a / (1.0f + __expf(-a));          // silu
  xs[(size_t)b * DI + d]   = s;
  xs16[(size_t)b * DI + d] = (u16)bf16_rne(s);
  float* so = st_out + sbase;
  so[0] = c1; so[1] = c2; so[2] = c3; so[3] = xc;
}

// ---------------------------------------------------------------------------
// selective-scan update (16 states/channel) + gating; new SSM state -> d_out
// ---------------------------------------------------------------------------
__global__ __launch_bounds__(256)
void ssm_kernel(const float* __restrict__ st_in, const float* __restrict__ xz,
                const float* __restrict__ xdb, const float* __restrict__ dt,
                const float* __restrict__ xs, const float* __restrict__ A_log,
                const float* __restrict__ Dp, float* __restrict__ st_out,
                u16* __restrict__ y16, int l)
{
  const int idx = blockIdx.x * 256 + threadIdx.x;   // < NB*DI
  const int b = idx >> 11;
  const int d = idx & (DI - 1);
  const size_t sbase = ((size_t)(l * NB + b) * DI + d) * (DCV + DSS);
  const float dtv = dt[(size_t)b * DI + d];
  const float xsv = xs[(size_t)b * DI + d];
  const float zv  = xz[(size_t)b * (2 * DI) + DI + d];
  const float dtx = dtv * xsv;
  const float* al = A_log + ((size_t)l * DI + d) * DSS;
  const float* Bm = xdb + (size_t)b * XDBW + DTR;
  const float* Cm = Bm + DSS;
  float acc = 0.0f;
#pragma unroll
  for (int n = 0; n < DSS; ++n) {
    const float Aa = -__expf(al[n]);
    const float ss = st_in[sbase + DCV + n] * __expf(dtv * Aa) + dtx * Bm[n];
    st_out[sbase + DCV + n] = ss;
    acc += ss * Cm[n];
  }
  acc += Dp[l * DI + d] * xsv;
  acc *= zv / (1.0f + __expf(-zv));                 // * silu(z)
  y16[(size_t)b * DI + d] = (u16)bf16_rne(acc);
}

// ---------------------------------------------------------------------------
extern "C" void kernel_launch(void* const* d_in, const int* in_sizes, int n_in,
                              void* d_out, int out_size, void* d_ws, size_t ws_size,
                              hipStream_t stream)
{
  const float* x        = (const float*)d_in[0];
  const float* rnn      = (const float*)d_in[1];
  const float* W_in     = (const float*)d_in[2];
  const float* b_in     = (const float*)d_in[3];
  const float* W_out    = (const float*)d_in[4];
  const float* b_out    = (const float*)d_in[5];
  const float* in_proj  = (const float*)d_in[6];
  const float* conv_W   = (const float*)d_in[7];
  const float* conv_b   = (const float*)d_in[8];
  const float* x_proj   = (const float*)d_in[9];
  const float* dt_proj  = (const float*)d_in[10];
  const float* dt_b     = (const float*)d_in[11];
  const float* A_log    = (const float*)d_in[12];
  const float* Dp       = (const float*)d_in[13];
  const float* out_proj = (const float*)d_in[14];
  const float* norm_w   = (const float*)d_in[15];
  const float* norm_b   = (const float*)d_in[16];
  const float* normf_w  = (const float*)d_in[17];
  const float* normf_b  = (const float*)d_in[18];

  float* out    = (float*)d_out;                 // (1, B, OUT)
  float* st_out = out + (size_t)NB * DOUT;       // (L, B, DI*20)

  // ---- workspace carve-up ----
  float* h   = (float*)d_ws;                     // B x DM
  float* res = h   + (size_t)NB * DM;            // B x DM
  float* xz  = res + (size_t)NB * DM;            // B x 2*DI
  float* xs  = xz  + (size_t)NB * 2 * DI;        // B x DI
  float* xdb = xs  + (size_t)NB * DI;            // B x 96
  float* dtb = xdb + (size_t)NB * XDBW;          // B x DI
  u16* b16   = (u16*)(dtb + (size_t)NB * DI);    // bf16 region
  u16* x16   = b16;                     b16 += (size_t)NB * DIN;
  u16* hn16  = b16;                     b16 += (size_t)NB * DM;
  u16* xs16  = b16;                     b16 += (size_t)NB * DI;
  u16* xdb16 = b16;                     b16 += (size_t)NB * XDBW;
  u16* yb16  = b16;                     b16 += (size_t)NB * DI;
  u16* Wi16  = b16;                     b16 += (size_t)DM * DIN;
  u16* Wo16  = b16;                     b16 += (size_t)DOUT * DM;
  u16* ip16  = b16;                     b16 += (size_t)NL * 2 * DI * DM;
  u16* xp16  = b16;                     b16 += (size_t)NL * XDBW * DI;
  u16* dtp16 = b16;                     b16 += (size_t)NL * DI * DTR;
  u16* op16  = b16;                     b16 += (size_t)NL * DM * DI;

  const dim3 blk(256);
  auto cvt = [&](const float* s, u16* d, size_t n) {
    const long n8 = (long)(n >> 3);
    cvt_bf16_kernel<<<(unsigned)((n8 + 255) / 256), blk, 0, stream>>>(s, (unsigned*)d, n8);
  };
  auto gemm = [&](const u16* A, const u16* W, const float* bias, float* C,
                  u16* C16, int ldc16, int M, int N, int K,
                  int lda, int ldb, int ldc, int epi) {
    dim3 grid((N + 127) / 128, (M + 127) / 128);
    wmma_gemm_kernel<<<grid, blk, 0, stream>>>(A, W, bias, C, C16, ldc16,
                                               M, N, K, lda, ldb, ldc, epi);
  };

  // ---- one-shot f32 -> bf16 conversion of GEMM operands ----
  cvt(x,        x16,  (size_t)NB * DIN);
  cvt(W_in,     Wi16, (size_t)DM * DIN);
  cvt(W_out,    Wo16, (size_t)DOUT * DM);
  cvt(in_proj,  ip16, (size_t)NL * 2 * DI * DM);
  cvt(x_proj,   xp16, (size_t)NL * XDBW * DI);
  cvt(dt_proj,  dtp16,(size_t)NL * DI * DTR);
  cvt(out_proj, op16, (size_t)NL * DM * DI);

  // h = x @ W_in^T + b_in
  gemm(x16, Wi16, b_in, h, nullptr, 0, NB, DM, DIN, DIN, DIN, DM, EPI_BIAS);

  for (int l = 0; l < NL; ++l) {
    ln_kernel<<<NB, blk, 0, stream>>>(h, res, hn16, norm_w + l * DM, norm_b + l * DM,
                                      l > 0 ? 1 : 0);
    gemm(hn16, ip16 + (size_t)l * 2 * DI * DM, nullptr, xz, nullptr, 0,
         NB, 2 * DI, DM, DM, DM, 2 * DI, EPI_NONE);
    conv_kernel<<<(NB * DI) / 256, blk, 0, stream>>>(rnn, xz, conv_W, conv_b,
                                                     st_out, xs, xs16, l);
    gemm(xs16, xp16 + (size_t)l * XDBW * DI, nullptr, xdb, xdb16, XDBW,
         NB, XDBW, DI, DI, DI, XDBW, EPI_NONE);
    gemm(xdb16, dtp16 + (size_t)l * DI * DTR, dt_b + l * DI, dtb, nullptr, 0,
         NB, DI, DTR, XDBW, DTR, DI, EPI_BIAS_SOFTPLUS);
    ssm_kernel<<<(NB * DI) / 256, blk, 0, stream>>>(rnn, xz, xdb, dtb, xs, A_log, Dp,
                                                    st_out, yb16, l);
    gemm(yb16, op16 + (size_t)l * DM * DI, nullptr, h, nullptr, 0,
         NB, DM, DI, DI, DI, DM, EPI_NONE);
  }

  ln_kernel<<<NB, blk, 0, stream>>>(h, res, hn16, normf_w, normf_b, 1);
  gemm(hn16, Wo16, b_out, out, nullptr, 0, NB, DOUT, DM, DM, DM, DOUT, EPI_BIAS);
}
